// BUSLSTMCell_88519275971256
// MI455X (gfx1250) — compile-verified, matrix operand order checked by reading, not run on hardware
//
#include <hip/hip_runtime.h>
#include <hip/hip_bf16.h>

// CDNA5 / gfx1250: wave32, WMMA 16x16x32 bf16 -> f32 accumulate,
// async global->LDS staging (ASYNCcnt) via global_load_async_to_lds_b128.
typedef __attribute__((ext_vector_type(16))) __bf16 v16bf;
typedef __attribute__((ext_vector_type(8)))  float  v8f;
typedef __attribute__((ext_vector_type(4)))  __bf16 v4bf;
typedef int v4i_ld __attribute__((vector_size(4 * sizeof(int))));

#define BM 128
#define BN 64
#define BK 64
#define LDST 80           // LDS row stride (bf16 elems): 64 + 16 pad -> 160B rows, 32B-aligned frags
#define NROWS 8192
#define DD 1024

#if defined(__has_builtin)
#if __has_builtin(__builtin_amdgcn_global_load_async_to_lds_b128)
#define HAVE_ASYNC_LDS 1
#else
#define HAVE_ASYNC_LDS 0
#endif
#else
#define HAVE_ASYNC_LDS 0
#endif

// ---------- helpers ----------
__device__ __forceinline__ __bf16 tobf(float f) {
  unsigned u = __float_as_uint(f);
  u += 0x7FFFu + ((u >> 16) & 1u);          // round-to-nearest-even
  unsigned short h = (unsigned short)(u >> 16);
  return __builtin_bit_cast(__bf16, h);
}

__device__ __forceinline__ float sigmoidf_(float x) {
  return 1.0f / (1.0f + __expf(-x));
}

__device__ __forceinline__ v8f wmma_bf16(v16bf a, v16bf b, v8f c) {
  return __builtin_amdgcn_wmma_f32_16x16x32_bf16(
      /*neg_a=*/false, a, /*neg_b=*/false, b,
      /*c_mod=*/(short)0, c, /*reuse_a=*/false, /*reuse_b=*/false);
}

#if HAVE_ASYNC_LDS
__device__ __forceinline__ void async_cp16(const __bf16* g, __bf16* l) {
  __builtin_amdgcn_global_load_async_to_lds_b128(
      (__attribute__((address_space(1))) v4i_ld*)g,
      (__attribute__((address_space(3))) v4i_ld*)l, 0, 0);
}
#endif

__device__ __forceinline__ void wait_async() {
#if HAVE_ASYNC_LDS
#if __has_builtin(__builtin_amdgcn_s_wait_asynccnt)
  __builtin_amdgcn_s_wait_asynccnt(0);
#else
  asm volatile("s_wait_asynccnt 0x0" ::: "memory");
#endif
#endif
}

// Stage one 64x64 bf16 tile (row-major, leading dim ld) global -> LDS.
// 256 threads: thread t covers row t>>2, 16 contiguous elems at col (t&3)*16.
__device__ __forceinline__ void stage64(const __bf16* __restrict__ src, int ld,
                                        __bf16* __restrict__ lds, int tid) {
  int r  = tid >> 2;
  int c0 = (tid & 3) << 4;
  const __bf16* p = src + (size_t)r * ld + c0;
  __bf16* q = lds + r * LDST + c0;
#if HAVE_ASYNC_LDS
  async_cp16(p, q);           // global_load_async_to_lds_b128 x2
  async_cp16(p + 8, q + 8);
#else
  ((uint4*)q)[0] = ((const uint4*)p)[0];
  ((uint4*)q)[1] = ((const uint4*)p)[1];
#endif
}

// WMMA bf16 operand fragment from an LDS tile (row-major, stride LDST).
__device__ __forceinline__ v16bf frag(const __bf16* __restrict__ lds, int row0,
                                      int kbase, int lane) {
  const __bf16* p = lds + (row0 + (lane & 15)) * LDST + kbase + ((lane >> 4) << 4);
  return *(const v16bf*)p;   // 32B, 32B-aligned -> ds_load_b128 x2
}

// ---------- f32 -> bf16 conversion ----------
__global__ void __launch_bounds__(256)
cvt_f32_bf16(const float* __restrict__ s, __bf16* __restrict__ d, int n4) {
  int i = blockIdx.x * 256 + threadIdx.x;
  if (i < n4) {
    float4 v = ((const float4*)s)[i];
    v4bf b;
    b.x = tobf(v.x); b.y = tobf(v.y); b.z = tobf(v.z); b.w = tobf(v.w);
    ((v4bf*)d)[i] = b;
  }
}

// ---------- phase 1: z gate + x_t blend (double-buffered async pipeline) ----------
// z = sigmoid([x_l,x_r] @ W_z^T + b_z); x_t = z*x_l + (1-z)*x_r
__global__ void __launch_bounds__(256)
zgate_kernel(const __bf16* __restrict__ xl_bf, const __bf16* __restrict__ xr_bf,
             const __bf16* __restrict__ wz,    // [1024, 2048] bf16
             const float* __restrict__ b_z,
             const float* __restrict__ x_l, const float* __restrict__ x_r,
             float* __restrict__ out_x, __bf16* __restrict__ xt_bf) {
  __shared__ __align__(32) __bf16 sA[2][BM * LDST];   // 2 x 20 KB
  __shared__ __align__(32) __bf16 sB[2][BN * LDST];   // 2 x 10 KB
  int tid = threadIdx.x, lane = tid & 31, wid = tid >> 5;
  int wm = (wid & 3) << 5, wn = (wid >> 2) << 5;      // 32x32 wave tile, 4x2 wave grid
  int m0 = blockIdx.y * BM, n0 = blockIdx.x * BN;
  v8f acc[2][2] = {};
  const int NST = 2048 / BK;
  auto stage = [&](int s, int b) {
    int k = s * BK;
    const __bf16* abase = (k < 1024 ? xl_bf : xr_bf) + (size_t)m0 * DD + (k & 1023);
    stage64(abase, DD, sA[b], tid);
    stage64(abase + (size_t)64 * DD, DD, sA[b] + 64 * LDST, tid);
    stage64(wz + (size_t)n0 * 2048 + k, 2048, sB[b], tid);
  };
  stage(0, 0);
  for (int s = 0; s < NST; ++s) {
    int cur = s & 1;
    wait_async();
    __syncthreads();
    if (s + 1 < NST) stage(s + 1, cur ^ 1);   // DMA next stage under the WMMAs
    const __bf16* A = sA[cur];
    const __bf16* B = sB[cur];
#pragma unroll
    for (int kk = 0; kk < 2; ++kk) {
      v16bf a0 = frag(A, wm,      kk * 32, lane);
      v16bf a1 = frag(A, wm + 16, kk * 32, lane);
      v16bf b0 = frag(B, wn,      kk * 32, lane);
      v16bf b1 = frag(B, wn + 16, kk * 32, lane);
      acc[0][0] = wmma_bf16(a0, b0, acc[0][0]);
      acc[0][1] = wmma_bf16(a0, b1, acc[0][1]);
      acc[1][0] = wmma_bf16(a1, b0, acc[1][0]);
      acc[1][1] = wmma_bf16(a1, b1, acc[1][1]);
    }
  }
#pragma unroll
  for (int tm = 0; tm < 2; ++tm) {
    int mb = m0 + wm + tm * 16 + ((lane >> 4) << 3);
#pragma unroll
    for (int tn = 0; tn < 2; ++tn) {
      int n = n0 + wn + tn * 16 + (lane & 15);
      float bz = b_z[n];
#pragma unroll
      for (int r = 0; r < 8; ++r) {
        size_t id = (size_t)(mb + r) * DD + n;
        float z  = sigmoidf_(acc[tm][tn][r] + bz);
        float xt = z * x_l[id] + (1.0f - z) * x_r[id];
        out_x[id] = xt;
        xt_bf[id] = tobf(xt);
      }
    }
  }
}

// ---------- phase 2: x_in = x_t @ W_xin^T (no bias, double-buffered) ----------
__global__ void __launch_bounds__(256)
xin_kernel(const __bf16* __restrict__ xt_bf, const __bf16* __restrict__ wxin, // [4096,1024]
           float* __restrict__ xin_out) {                                     // [N,4096]
  __shared__ __align__(32) __bf16 sA[2][BM * LDST];
  __shared__ __align__(32) __bf16 sB[2][BN * LDST];
  int tid = threadIdx.x, lane = tid & 31, wid = tid >> 5;
  int wm = (wid & 3) << 5, wn = (wid >> 2) << 5;
  int m0 = blockIdx.y * BM, n0 = blockIdx.x * BN;
  v8f acc[2][2] = {};
  const int NST = 1024 / BK;
  auto stage = [&](int s, int b) {
    int k = s * BK;
    const __bf16* abase = xt_bf + (size_t)m0 * DD + k;
    stage64(abase, DD, sA[b], tid);
    stage64(abase + (size_t)64 * DD, DD, sA[b] + 64 * LDST, tid);
    stage64(wxin + (size_t)n0 * DD + k, DD, sB[b], tid);
  };
  stage(0, 0);
  for (int s = 0; s < NST; ++s) {
    int cur = s & 1;
    wait_async();
    __syncthreads();
    if (s + 1 < NST) stage(s + 1, cur ^ 1);
    const __bf16* A = sA[cur];
    const __bf16* B = sB[cur];
#pragma unroll
    for (int kk = 0; kk < 2; ++kk) {
      v16bf a0 = frag(A, wm,      kk * 32, lane);
      v16bf a1 = frag(A, wm + 16, kk * 32, lane);
      v16bf b0 = frag(B, wn,      kk * 32, lane);
      v16bf b1 = frag(B, wn + 16, kk * 32, lane);
      acc[0][0] = wmma_bf16(a0, b0, acc[0][0]);
      acc[0][1] = wmma_bf16(a0, b1, acc[0][1]);
      acc[1][0] = wmma_bf16(a1, b0, acc[1][0]);
      acc[1][1] = wmma_bf16(a1, b1, acc[1][1]);
    }
  }
#pragma unroll
  for (int tm = 0; tm < 2; ++tm) {
    int mb = m0 + wm + tm * 16 + ((lane >> 4) << 3);
#pragma unroll
    for (int tn = 0; tn < 2; ++tn) {
      int n = n0 + wn + tn * 16 + (lane & 15);
#pragma unroll
      for (int r = 0; r < 8; ++r)
        xin_out[(size_t)(mb + r) * 4096 + n] = acc[tm][tn][r];
    }
  }
}

// ---------- phase 3: fused i / f_l / f_r / g gates + c_t ----------
// A = [h_l,h_r,c_l,c_r] (K=4096) staged once, reused for W_i/W_fl/W_fr (+W_g for K<2048).
__global__ void __launch_bounds__(256)
gates_kernel(const __bf16* __restrict__ hl, const __bf16* __restrict__ hr,
             const __bf16* __restrict__ cl, const __bf16* __restrict__ cr,
             const __bf16* __restrict__ wi, const __bf16* __restrict__ wfl,
             const __bf16* __restrict__ wfr, const __bf16* __restrict__ wg,
             const float* __restrict__ b_i, const float* __restrict__ b_fl,
             const float* __restrict__ b_fr, const float* __restrict__ b_g,
             const float* __restrict__ xin, const float* __restrict__ c_l,
             const float* __restrict__ c_r, float* __restrict__ out_c,
             __bf16* __restrict__ ct_bf) {
  __shared__ __align__(32) __bf16 sA [BM * LDST];     // 20 KB
  __shared__ __align__(32) __bf16 sBi[BN * LDST];     // 4 x 10 KB
  __shared__ __align__(32) __bf16 sBl[BN * LDST];
  __shared__ __align__(32) __bf16 sBr[BN * LDST];
  __shared__ __align__(32) __bf16 sBg[BN * LDST];
  int tid = threadIdx.x, lane = tid & 31, wid = tid >> 5;
  int wm = (wid & 3) << 5, wn = (wid >> 2) << 5;
  int m0 = blockIdx.y * BM, n0 = blockIdx.x * BN;
  v8f ai[2][2] = {}, afl[2][2] = {}, afr[2][2] = {}, ag[2][2] = {};
  for (int k = 0; k < 4096; k += BK) {
    const __bf16* abase =
        (k < 1024 ? hl : k < 2048 ? hr : k < 3072 ? cl : cr) +
        (size_t)m0 * DD + (k & 1023);
    bool do_g = (k < 2048);
    __syncthreads();                              // previous stage fully consumed
    stage64(abase, DD, sA, tid);
    stage64(abase + (size_t)64 * DD, DD, sA + 64 * LDST, tid);
    stage64(wi  + (size_t)n0 * 4096 + k, 4096, sBi, tid);
    stage64(wfl + (size_t)n0 * 4096 + k, 4096, sBl, tid);
    stage64(wfr + (size_t)n0 * 4096 + k, 4096, sBr, tid);
    if (do_g) stage64(wg + (size_t)n0 * 2048 + k, 2048, sBg, tid);
    wait_async();
    __syncthreads();
#pragma unroll
    for (int kk = 0; kk < 2; ++kk) {
      v16bf a0 = frag(sA, wm,      kk * 32, lane);
      v16bf a1 = frag(sA, wm + 16, kk * 32, lane);
#pragma unroll
      for (int tn = 0; tn < 2; ++tn) {
        v16bf bi = frag(sBi, wn + tn * 16, kk * 32, lane);
        ai[0][tn]  = wmma_bf16(a0, bi, ai[0][tn]);
        ai[1][tn]  = wmma_bf16(a1, bi, ai[1][tn]);
        v16bf bl = frag(sBl, wn + tn * 16, kk * 32, lane);
        afl[0][tn] = wmma_bf16(a0, bl, afl[0][tn]);
        afl[1][tn] = wmma_bf16(a1, bl, afl[1][tn]);
        v16bf br = frag(sBr, wn + tn * 16, kk * 32, lane);
        afr[0][tn] = wmma_bf16(a0, br, afr[0][tn]);
        afr[1][tn] = wmma_bf16(a1, br, afr[1][tn]);
        if (do_g) {
          v16bf bg = frag(sBg, wn + tn * 16, kk * 32, lane);
          ag[0][tn] = wmma_bf16(a0, bg, ag[0][tn]);
          ag[1][tn] = wmma_bf16(a1, bg, ag[1][tn]);
        }
      }
    }
  }
#pragma unroll
  for (int tm = 0; tm < 2; ++tm) {
    int mb = m0 + wm + tm * 16 + ((lane >> 4) << 3);
#pragma unroll
    for (int tn = 0; tn < 2; ++tn) {
      int n = n0 + wn + tn * 16 + (lane & 15);
      float bi = b_i[n], bl = b_fl[n], br = b_fr[n], bg = b_g[n];
#pragma unroll
      for (int r = 0; r < 8; ++r) {
        int m = mb + r;
        size_t ix = (size_t)m * 4096;
        size_t id = (size_t)m * DD + n;
        float i_t = sigmoidf_(ai[tm][tn][r]  + bi + xin[ix + n]);        // + x_i
        float xf  = xin[ix + 1024 + n];                                   // x_f (shared)
        float flt = sigmoidf_(afl[tm][tn][r] + bl + xf);
        float frt = sigmoidf_(afr[tm][tn][r] + br + xf);
        float g_t = tanhf(ag[tm][tn][r] + bg + xin[ix + 3072 + n]);       // + x_g
        float ct  = flt * c_l[id] + frt * c_r[id] + i_t * g_t;
        out_c[id] = ct;
        ct_bf[id] = tobf(ct);
      }
    }
  }
}

// ---------- phase 4: o gate + h_t (double-buffered) ----------
// o = sigmoid([h_l,h_r,c_t] @ W_o^T + b_o + x_o); h = o * tanh(c_t)
__global__ void __launch_bounds__(256)
ogate_kernel(const __bf16* __restrict__ hl, const __bf16* __restrict__ hr,
             const __bf16* __restrict__ ctb, const __bf16* __restrict__ wo, // [1024,3072]
             const float* __restrict__ b_o, const float* __restrict__ xin,
             const float* __restrict__ c_t, float* __restrict__ out_h) {
  __shared__ __align__(32) __bf16 sA[2][BM * LDST];
  __shared__ __align__(32) __bf16 sB[2][BN * LDST];
  int tid = threadIdx.x, lane = tid & 31, wid = tid >> 5;
  int wm = (wid & 3) << 5, wn = (wid >> 2) << 5;
  int m0 = blockIdx.y * BM, n0 = blockIdx.x * BN;
  v8f acc[2][2] = {};
  const int NST = 3072 / BK;
  auto stage = [&](int s, int b) {
    int k = s * BK;
    const __bf16* abase =
        (k < 1024 ? hl : k < 2048 ? hr : ctb) + (size_t)m0 * DD + (k & 1023);
    stage64(abase, DD, sA[b], tid);
    stage64(abase + (size_t)64 * DD, DD, sA[b] + 64 * LDST, tid);
    stage64(wo + (size_t)n0 * 3072 + k, 3072, sB[b], tid);
  };
  stage(0, 0);
  for (int s = 0; s < NST; ++s) {
    int cur = s & 1;
    wait_async();
    __syncthreads();
    if (s + 1 < NST) stage(s + 1, cur ^ 1);
    const __bf16* A = sA[cur];
    const __bf16* B = sB[cur];
#pragma unroll
    for (int kk = 0; kk < 2; ++kk) {
      v16bf a0 = frag(A, wm,      kk * 32, lane);
      v16bf a1 = frag(A, wm + 16, kk * 32, lane);
      v16bf b0 = frag(B, wn,      kk * 32, lane);
      v16bf b1 = frag(B, wn + 16, kk * 32, lane);
      acc[0][0] = wmma_bf16(a0, b0, acc[0][0]);
      acc[0][1] = wmma_bf16(a0, b1, acc[0][1]);
      acc[1][0] = wmma_bf16(a1, b0, acc[1][0]);
      acc[1][1] = wmma_bf16(a1, b1, acc[1][1]);
    }
  }
#pragma unroll
  for (int tm = 0; tm < 2; ++tm) {
    int mb = m0 + wm + tm * 16 + ((lane >> 4) << 3);
#pragma unroll
    for (int tn = 0; tn < 2; ++tn) {
      int n = n0 + wn + tn * 16 + (lane & 15);
      float bo = b_o[n];
#pragma unroll
      for (int r = 0; r < 8; ++r) {
        int m = mb + r;
        size_t id = (size_t)m * DD + n;
        float o = sigmoidf_(acc[tm][tn][r] + bo + xin[(size_t)m * 4096 + 2048 + n]);
        out_h[id] = o * tanhf(c_t[id]);
      }
    }
  }
}

// ---------- host ----------
extern "C" void kernel_launch(void* const* d_in, const int* in_sizes, int n_in,
                              void* d_out, int out_size, void* d_ws, size_t ws_size,
                              hipStream_t stream) {
  (void)in_sizes; (void)n_in; (void)out_size; (void)ws_size;
  const float* x_l  = (const float*)d_in[0];
  const float* h_l  = (const float*)d_in[1];
  const float* c_l  = (const float*)d_in[2];
  const float* x_r  = (const float*)d_in[3];
  const float* h_r  = (const float*)d_in[4];
  const float* c_r  = (const float*)d_in[5];
  const float* W_i  = (const float*)d_in[6];
  const float* b_i  = (const float*)d_in[7];
  const float* W_fl = (const float*)d_in[8];
  const float* b_fl = (const float*)d_in[9];
  const float* W_fr = (const float*)d_in[10];
  const float* b_fr = (const float*)d_in[11];
  const float* W_xin= (const float*)d_in[12];
  const float* W_o  = (const float*)d_in[13];
  const float* b_o  = (const float*)d_in[14];
  const float* W_z  = (const float*)d_in[15];
  const float* b_z  = (const float*)d_in[16];
  const float* W_g  = (const float*)d_in[17];
  const float* b_g  = (const float*)d_in[18];
  // d_in[19] = dim (always 1)

  const size_t NE = (size_t)NROWS * DD;

  // workspace layout (bf16 copies + f32 x_in scratch), ~302 MB
  __bf16* xl_bf  = (__bf16*)d_ws;
  __bf16* xr_bf  = xl_bf + NE;
  __bf16* hl_bf  = xr_bf + NE;
  __bf16* hr_bf  = hl_bf + NE;
  __bf16* cl_bf  = hr_bf + NE;
  __bf16* cr_bf  = cl_bf + NE;
  __bf16* xt_bf  = cr_bf + NE;
  __bf16* ct_bf  = xt_bf + NE;
  __bf16* wz_bf  = ct_bf + NE;
  __bf16* wxin_bf= wz_bf  + (size_t)1024 * 2048;
  __bf16* wi_bf  = wxin_bf+ (size_t)4096 * 1024;
  __bf16* wfl_bf = wi_bf  + (size_t)1024 * 4096;
  __bf16* wfr_bf = wfl_bf + (size_t)1024 * 4096;
  __bf16* wg_bf  = wfr_bf + (size_t)1024 * 4096;
  __bf16* wo_bf  = wg_bf  + (size_t)1024 * 2048;
  float*  xin_f  = (float*)(wo_bf + (size_t)1024 * 3072);

  float* out_x = (float*)d_out;
  float* out_h = out_x + NE;
  float* out_c = out_h + NE;

  dim3 blk(256);
  auto cv = [&](const float* s, __bf16* d, size_t n) {
    int n4 = (int)(n >> 2);
    cvt_f32_bf16<<<dim3((unsigned)((n4 + 255) / 256)), blk, 0, stream>>>(s, d, n4);
  };
  cv(x_l, xl_bf, NE);  cv(x_r, xr_bf, NE);
  cv(h_l, hl_bf, NE);  cv(h_r, hr_bf, NE);
  cv(c_l, cl_bf, NE);  cv(c_r, cr_bf, NE);
  cv(W_z,  wz_bf,  (size_t)1024 * 2048);
  cv(W_xin,wxin_bf,(size_t)4096 * 1024);
  cv(W_i,  wi_bf,  (size_t)1024 * 4096);
  cv(W_fl, wfl_bf, (size_t)1024 * 4096);
  cv(W_fr, wfr_bf, (size_t)1024 * 4096);
  cv(W_g,  wg_bf,  (size_t)1024 * 2048);
  cv(W_o,  wo_bf,  (size_t)1024 * 3072);

  dim3 gridD(DD / BN, NROWS / BM);          // 16 x 64
  dim3 grid4(4096 / BN, NROWS / BM);        // 64 x 64

  zgate_kernel<<<gridD, blk, 0, stream>>>(xl_bf, xr_bf, wz_bf, b_z, x_l, x_r,
                                          out_x, xt_bf);
  xin_kernel<<<grid4, blk, 0, stream>>>(xt_bf, wxin_bf, xin_f);
  gates_kernel<<<gridD, blk, 0, stream>>>(hl_bf, hr_bf, cl_bf, cr_bf,
                                          wi_bf, wfl_bf, wfr_bf, wg_bf,
                                          b_i, b_fl, b_fr, b_g,
                                          xin_f, c_l, c_r, out_c, ct_bf);
  ogate_kernel<<<gridD, blk, 0, stream>>>(hl_bf, hr_bf, ct_bf, wo_bf,
                                          b_o, xin_f, out_c, out_h);
}
